// Attention_4853313045001
// MI455X (gfx1250) — compile-verified, matrix operand order checked by reading, not run on hardware
//
#include <hip/hip_runtime.h>
#include <cstdint>

// ---------------------------------------------------------------------------
// Multi-head attention forward for MI455X (gfx1250).
// bf16 WMMA (v_wmma_f32_16x16x32_bf16) + f32 accumulation everywhere,
// async global->LDS staging (global_load_async_to_lds_b128, ASYNCcnt).
//
// Pipeline:
//   0a. x fp32 -> bf16                     (xb [8192][768])
//   0b. w_qkv, w_proj fp32 -> bf16, transposed to [n][k]
//   1.  QKV GEMM  -> q,k as [bh][n][d] bf16, v as [bh][d][n] bf16 (pre-T)
//   2.  flash attention per (b,h)          -> attn [8192][768] bf16
//   3.  out = attn @ w_proj + b_proj       -> fp32
// ---------------------------------------------------------------------------

typedef __bf16 bf16_t;
typedef __attribute__((ext_vector_type(16))) __bf16 v16bf;
typedef __attribute__((ext_vector_type(8)))  __bf16 v8bf;
typedef __attribute__((ext_vector_type(4)))  __bf16 v4bf;
typedef __attribute__((ext_vector_type(8)))  float  v8f;

#define DIMK 768
#define NQKV 2304
#define SEQ  1024
#define NB   8
#define NH   12
#define HD   64
#define MTOT (NB * SEQ)   // 8192

#define GBM 128
#define GBN 128
#define GBK 64
#define LDT 72            // padded LDS row (bf16): 144 B -> rows 16B-aligned

// ---- CDNA5 async global->LDS copy (16 bytes per lane, ASYNCcnt-tracked) ----
__device__ __forceinline__ void asyncCopy16(bf16_t* ldsDst, const bf16_t* gSrc) {
    uint32_t l = (uint32_t)(uintptr_t)ldsDst;   // low 32 bits == LDS byte offset
    asm volatile("global_load_async_to_lds_b128 %0, %1, off"
                 :: "v"(l), "v"(gSrc) : "memory");
}
__device__ __forceinline__ void asyncWait0() {
    asm volatile("s_wait_asynccnt 0x0" ::: "memory");
}

// Load one 16x32 bf16 WMMA fragment (A layout; B mirrors it for [n][k] tiles).
// ISA 7.12.2: lanes 0-15 -> row M=lane, K={0..7,16..23};
//             lanes 16-31 -> row M=lane-16, K={8..15,24..31}.
__device__ __forceinline__ v16bf ldsFrag(const bf16_t* __restrict__ base, int ld) {
    const int lane = threadIdx.x & 31;
    const int grp  = lane >> 4;
    const int r    = lane & 15;
    const bf16_t* p = base + r * ld + grp * 8;
    v8bf lo = *(const v8bf*)(p);
    v8bf hi = *(const v8bf*)(p + 16);
    v16bf f;
#pragma unroll
    for (int j = 0; j < 8; ++j) { f[j] = lo[j]; f[j + 8] = hi[j]; }
    return f;
}

__device__ __forceinline__ v8f wmma_bf16(v16bf a, v16bf b, v8f c) {
    return __builtin_amdgcn_wmma_f32_16x16x32_bf16(
        false, a, false, b, (short)0, c, false, false);
}

// ---------------------------------------------------------------------------
// Kernel 0a: x fp32 -> bf16 (vectorized x4)
// ---------------------------------------------------------------------------
__global__ void __launch_bounds__(256) cvt_x_kernel(
    const float* __restrict__ X, bf16_t* __restrict__ Xb)
{
    size_t i = ((size_t)blockIdx.x * 256 + threadIdx.x) * 4;
    float4 v = *(const float4*)(X + i);
    v4bf t; t[0] = (bf16_t)v.x; t[1] = (bf16_t)v.y;
            t[2] = (bf16_t)v.z; t[3] = (bf16_t)v.w;
    *(v4bf*)(Xb + i) = t;
}

// ---------------------------------------------------------------------------
// Kernel 0b: W[k][n] fp32 -> Wt[n][k] bf16  (coalesced reads)
// ---------------------------------------------------------------------------
__global__ void __launch_bounds__(256) transpose_w_kernel(
    const float* __restrict__ W, bf16_t* __restrict__ Wt, int K, int N)
{
    int id = blockIdx.x * 256 + threadIdx.x;
    if (id >= K * N) return;
    int k = id / N, n = id - k * N;
    Wt[(size_t)n * K + k] = (bf16_t)W[id];
}

// ---------------------------------------------------------------------------
// Kernel 1: QKV GEMM  xb[8192][768] @ wqkvT -> q,k:[bh][n][d]  v:[bh][d][n]
// 128x128x64 tiles, 8 waves, each wave 32x64 (2x4 frags, 8 WMMAs / k-step).
// ---------------------------------------------------------------------------
__global__ void __launch_bounds__(256) qkv_gemm_kernel(
    const bf16_t* __restrict__ Xb, const bf16_t* __restrict__ WqkvT,
    bf16_t* __restrict__ qg, bf16_t* __restrict__ kg, bf16_t* __restrict__ vtg)
{
    __shared__ bf16_t As[GBM * LDT];   // [m][k]
    __shared__ bf16_t Bs[GBN * LDT];   // [n][k]

    const int tid  = threadIdx.x;
    const int wave = tid >> 5;
    const int lane = tid & 31;
    const int grp  = lane >> 4;
    const int n16  = lane & 15;
    const int rowBase = blockIdx.x * GBM;
    const int colBase = blockIdx.y * GBN;
    const int m0 = (wave & 3) * 32;
    const int n0 = (wave >> 2) * 64;

    v8f acc[2][4];
    v8f zero = {};
#pragma unroll
    for (int i = 0; i < 2; ++i)
#pragma unroll
        for (int j = 0; j < 4; ++j) acc[i][j] = zero;

    for (int k0 = 0; k0 < DIMK; k0 += GBK) {
        __syncthreads();
        for (int i = tid; i < GBM * 8; i += 256) {      // 16B chunks per row
            int r = i >> 3, c = (i & 7) << 3;
            asyncCopy16(&As[r * LDT + c], Xb    + (size_t)(rowBase + r) * DIMK + k0 + c);
            asyncCopy16(&Bs[r * LDT + c], WqkvT + (size_t)(colBase + r) * DIMK + k0 + c);
        }
        asyncWait0();
        __syncthreads();
#pragma unroll
        for (int kk = 0; kk < GBK; kk += 32) {
            v16bf a0 = ldsFrag(&As[(m0)      * LDT + kk], LDT);
            v16bf a1 = ldsFrag(&As[(m0 + 16) * LDT + kk], LDT);
#pragma unroll
            for (int j = 0; j < 4; ++j) {
                v16bf b = ldsFrag(&Bs[(n0 + j * 16) * LDT + kk], LDT);
                acc[0][j] = wmma_bf16(a0, b, acc[0][j]);
                acc[1][j] = wmma_bf16(a1, b, acc[1][j]);
            }
        }
    }

    // scatter: col -> (s,h,d); v written pre-transposed [bh][d][n]
#pragma unroll
    for (int mi = 0; mi < 2; ++mi)
#pragma unroll
        for (int j = 0; j < 4; ++j) {
            int col  = colBase + n0 + j * 16 + n16;    // 0..2303
            int sIdx = col / DIMK;
            int rem  = col - sIdx * DIMK;
            int h    = rem >> 6;
            int d    = rem & 63;
#pragma unroll
            for (int r = 0; r < 8; ++r) {
                int row = rowBase + m0 + mi * 16 + r + grp * 8;  // 0..8191
                int b   = row >> 10;
                int n   = row & 1023;
                int bh  = b * NH + h;
                bf16_t val = (bf16_t)acc[mi][j][r];
                if (sIdx == 0)      qg [((size_t)bh * SEQ + n) * HD + d] = val;
                else if (sIdx == 1) kg [((size_t)bh * SEQ + n) * HD + d] = val;
                else                vtg[((size_t)bh * HD  + d) * SEQ + n] = val;
            }
        }
}

// ---------------------------------------------------------------------------
// Kernel 2: flash attention per (b,h); 64 query rows / block, 4 waves.
// ---------------------------------------------------------------------------
__global__ void __launch_bounds__(128) flash_attn_kernel(
    const bf16_t* __restrict__ qg, const bf16_t* __restrict__ kg,
    const bf16_t* __restrict__ vtg, bf16_t* __restrict__ attn)
{
    __shared__ bf16_t Qs[64 * LDT];            // [n][d]
    __shared__ bf16_t Ks[64 * LDT];            // [j][d]
    __shared__ bf16_t Vts[64 * LDT];           // [d][j]
    __shared__ bf16_t Ps[4 * 16 * LDT];        // per-wave P tiles [i][j]

    const int tid  = threadIdx.x;
    const int wave = tid >> 5;
    const int lane = tid & 31;
    const int grp  = lane >> 4;
    const int n16  = lane & 15;

    const int bh    = blockIdx.y;
    const int qbase = blockIdx.x * 64;

    const bf16_t* Qg = qg  + (size_t)bh * SEQ * HD;
    const bf16_t* Kg = kg  + (size_t)bh * SEQ * HD;
    const bf16_t* Vg = vtg + (size_t)bh * HD * SEQ;   // [d][n]

    // stage Q tile once (async), then keep its fragments in registers
    for (int i = tid; i < 64 * 8; i += 128) {
        int r = i >> 3, c = (i & 7) << 3;
        asyncCopy16(&Qs[r * LDT + c], Qg + (size_t)(qbase + r) * HD + c);
    }
    asyncWait0();
    __syncthreads();

    const int m0 = wave * 16;
    bf16_t* Pw = &Ps[wave * 16 * LDT];
    v16bf qa0 = ldsFrag(&Qs[m0 * LDT + 0], LDT);
    v16bf qa1 = ldsFrag(&Qs[m0 * LDT + 32], LDT);

    float mrun[8], lrun[8];
    v8f   oacc[4];
    v8f   zero = {};
#pragma unroll
    for (int r = 0; r < 8; ++r) { mrun[r] = -3.0e38f; lrun[r] = 0.0f; }
#pragma unroll
    for (int dt = 0; dt < 4; ++dt) oacc[dt] = zero;

    for (int kt = 0; kt < SEQ; kt += 64) {
        __syncthreads();
        for (int i = tid; i < 64 * 8; i += 128) {   // K rows + pre-T V rows
            int r = i >> 3, c = (i & 7) << 3;
            asyncCopy16(&Ks [r * LDT + c], Kg + (size_t)(kt + r) * HD + c);
            asyncCopy16(&Vts[r * LDT + c], Vg + (size_t)r * SEQ + kt + c);
        }
        asyncWait0();
        __syncthreads();

        // S = scale * Q @ K^T  (16 x 64 per wave)
        v8f s[4];
#pragma unroll
        for (int jt = 0; jt < 4; ++jt) {
            v16bf b0 = ldsFrag(&Ks[(jt * 16) * LDT + 0], LDT);
            v16bf b1 = ldsFrag(&Ks[(jt * 16) * LDT + 32], LDT);
            v8f t = zero;
            t = wmma_bf16(qa0, b0, t);
            t = wmma_bf16(qa1, b1, t);
#pragma unroll
            for (int r = 0; r < 8; ++r) t[r] *= 0.125f;   // HEAD_DIM^-0.5
            s[jt] = t;
        }

        // online softmax; lane owns rows r + 8*grp within its 16-lane half
        float mnew[8], alpha[8];
#pragma unroll
        for (int r = 0; r < 8; ++r) {
            float mx = fmaxf(fmaxf(s[0][r], s[1][r]), fmaxf(s[2][r], s[3][r]));
#pragma unroll
            for (int msk = 1; msk < 16; msk <<= 1)
                mx = fmaxf(mx, __shfl_xor(mx, msk, 32));
            float mn = fmaxf(mrun[r], mx);
            mnew[r]  = mn;
            alpha[r] = __expf(mrun[r] - mn);
            mrun[r]  = mn;
        }
#pragma unroll
        for (int jt = 0; jt < 4; ++jt)
#pragma unroll
            for (int r = 0; r < 8; ++r)
                s[jt][r] = __expf(s[jt][r] - mnew[r]);
#pragma unroll
        for (int r = 0; r < 8; ++r) {
            float sm = (s[0][r] + s[1][r]) + (s[2][r] + s[3][r]);
#pragma unroll
            for (int msk = 1; msk < 16; msk <<= 1)
                sm += __shfl_xor(sm, msk, 32);
            lrun[r] = lrun[r] * alpha[r] + sm;
        }
#pragma unroll
        for (int dt = 0; dt < 4; ++dt)
#pragma unroll
            for (int r = 0; r < 8; ++r) oacc[dt][r] *= alpha[r];

        // P -> LDS bf16 so it re-enters WMMA as an A fragment
#pragma unroll
        for (int jt = 0; jt < 4; ++jt)
#pragma unroll
            for (int r = 0; r < 8; ++r)
                Pw[(r + grp * 8) * LDT + jt * 16 + n16] = (bf16_t)s[jt][r];
        __syncthreads();

        v16bf pa0 = ldsFrag(&Pw[0], LDT);
        v16bf pa1 = ldsFrag(&Pw[32], LDT);
#pragma unroll
        for (int dt = 0; dt < 4; ++dt) {
            v16bf vb0 = ldsFrag(&Vts[(dt * 16) * LDT + 0], LDT);
            v16bf vb1 = ldsFrag(&Vts[(dt * 16) * LDT + 32], LDT);
            oacc[dt] = wmma_bf16(pa0, vb0, oacc[dt]);
            oacc[dt] = wmma_bf16(pa1, vb1, oacc[dt]);
        }
    }

    float inv[8];
#pragma unroll
    for (int r = 0; r < 8; ++r) inv[r] = 1.0f / lrun[r];
    const int b = bh / NH, h = bh - b * NH;
#pragma unroll
    for (int dt = 0; dt < 4; ++dt) {
        int col = h * HD + dt * 16 + n16;
#pragma unroll
        for (int r = 0; r < 8; ++r) {
            int n = qbase + m0 + r + grp * 8;
            attn[(size_t)(b * SEQ + n) * DIMK + col] = (bf16_t)(oacc[dt][r] * inv[r]);
        }
    }
}

// ---------------------------------------------------------------------------
// Kernel 3: out = attn @ w_proj + b_proj  (fp32 out), same tiling as kernel 1
// ---------------------------------------------------------------------------
__global__ void __launch_bounds__(256) proj_gemm_kernel(
    const bf16_t* __restrict__ Ag, const bf16_t* __restrict__ WpT,
    const float* __restrict__ bias, float* __restrict__ out)
{
    __shared__ bf16_t As[GBM * LDT];
    __shared__ bf16_t Bs[GBN * LDT];

    const int tid  = threadIdx.x;
    const int wave = tid >> 5;
    const int lane = tid & 31;
    const int grp  = lane >> 4;
    const int n16  = lane & 15;
    const int rowBase = blockIdx.x * GBM;
    const int colBase = blockIdx.y * GBN;
    const int m0 = (wave & 3) * 32;
    const int n0 = (wave >> 2) * 64;

    v8f acc[2][4];
    v8f zero = {};
#pragma unroll
    for (int i = 0; i < 2; ++i)
#pragma unroll
        for (int j = 0; j < 4; ++j) acc[i][j] = zero;

    for (int k0 = 0; k0 < DIMK; k0 += GBK) {
        __syncthreads();
        for (int i = tid; i < GBM * 8; i += 256) {
            int r = i >> 3, c = (i & 7) << 3;
            asyncCopy16(&As[r * LDT + c], Ag  + (size_t)(rowBase + r) * DIMK + k0 + c);
            asyncCopy16(&Bs[r * LDT + c], WpT + (size_t)(colBase + r) * DIMK + k0 + c);
        }
        asyncWait0();
        __syncthreads();
#pragma unroll
        for (int kk = 0; kk < GBK; kk += 32) {
            v16bf a0 = ldsFrag(&As[(m0)      * LDT + kk], LDT);
            v16bf a1 = ldsFrag(&As[(m0 + 16) * LDT + kk], LDT);
#pragma unroll
            for (int j = 0; j < 4; ++j) {
                v16bf b = ldsFrag(&Bs[(n0 + j * 16) * LDT + kk], LDT);
                acc[0][j] = wmma_bf16(a0, b, acc[0][j]);
                acc[1][j] = wmma_bf16(a1, b, acc[1][j]);
            }
        }
    }

#pragma unroll
    for (int mi = 0; mi < 2; ++mi)
#pragma unroll
        for (int j = 0; j < 4; ++j) {
            int col = colBase + n0 + j * 16 + n16;
            float bc = bias[col];
#pragma unroll
            for (int r = 0; r < 8; ++r) {
                int row = rowBase + m0 + mi * 16 + r + grp * 8;
                out[(size_t)row * DIMK + col] = acc[mi][j][r] + bc;
            }
        }
}

// ---------------------------------------------------------------------------
extern "C" void kernel_launch(void* const* d_in, const int* in_sizes, int n_in,
                              void* d_out, int out_size, void* d_ws, size_t ws_size,
                              hipStream_t stream) {
    (void)in_sizes; (void)n_in; (void)out_size; (void)ws_size;
    const float* x      = (const float*)d_in[0];
    const float* w_qkv  = (const float*)d_in[1];
    const float* w_proj = (const float*)d_in[2];
    const float* b_proj = (const float*)d_in[3];
    float* out = (float*)d_out;

    const size_t perT = (size_t)NB * NH * SEQ * HD;       // 6.29M elems
    bf16_t* qg     = (bf16_t*)d_ws;
    bf16_t* kgp    = qg + perT;
    bf16_t* vtg    = kgp + perT;
    bf16_t* attn   = vtg + perT;                          // [8192][768]
    bf16_t* xb     = attn + (size_t)MTOT * DIMK;          // [8192][768]
    bf16_t* wqkvT  = xb + (size_t)MTOT * DIMK;            // [2304][768]
    bf16_t* wprojT = wqkvT + (size_t)DIMK * NQKV;         // [768][768]

    cvt_x_kernel<<<(MTOT * DIMK) / (4 * 256), 256, 0, stream>>>(x, xb);
    transpose_w_kernel<<<(DIMK * NQKV) / 256, 256, 0, stream>>>(w_qkv, wqkvT, DIMK, NQKV);
    transpose_w_kernel<<<(DIMK * DIMK) / 256, 256, 0, stream>>>(w_proj, wprojT, DIMK, DIMK);

    dim3 g1(MTOT / GBM, NQKV / GBN);     // 64 x 18
    qkv_gemm_kernel<<<g1, 256, 0, stream>>>(xb, wqkvT, qg, kgp, vtg);

    dim3 g2(SEQ / 64, NB * NH);          // 16 x 96
    flash_attn_kernel<<<g2, 128, 0, stream>>>(qg, kgp, vtg, attn);

    dim3 g3(MTOT / GBM, DIMK / GBN);     // 64 x 6
    proj_gemm_kernel<<<g3, 256, 0, stream>>>(attn, wprojT, b_proj, out);
}
